// EfficientAttention_79903571574898
// MI455X (gfx1250) — compile-verified, matrix operand order checked by reading, not run on hardware
//
#include <hip/hip_runtime.h>
#include <hip/hip_bf16.h>

// ---------------------------------------------------------------------------
// EfficientAttention on MI455X (gfx1250, wave32, WMMA bf16, async LDS copies)
// ---------------------------------------------------------------------------

typedef __attribute__((ext_vector_type(16))) __bf16 v16bf;
typedef __attribute__((ext_vector_type(8)))  float  v8f;

union FragU {
    unsigned u[8];
    v16bf    v;
};

struct __align__(16) U4 { unsigned x, y, z, w; };

static __device__ __forceinline__ unsigned short f2bf(float f) {
    unsigned u = __builtin_bit_cast(unsigned, f);
    unsigned r = u + 0x7FFFu + ((u >> 16) & 1u);   // round-to-nearest-even
    return (unsigned short)(r >> 16);
}
static __device__ __forceinline__ float bf2f(unsigned short h) {
    unsigned u = ((unsigned)h) << 16;
    return __builtin_bit_cast(float, u);
}
static __device__ __forceinline__ v8f vzero8() {
    v8f z = {0.f, 0.f, 0.f, 0.f, 0.f, 0.f, 0.f, 0.f};
    return z;
}
// k index of dword-pair r for the 16-bit A/B fragment layout:
// lanes 0-15 hold K {0..7,16..23}, lanes 16-31 hold K {8..15,24..31}
static __device__ __forceinline__ int frag_k0(int r, int lane) {
    int kb = (lane >> 4) * 8;
    return (r < 4) ? (kb + 2 * r) : (16 + kb + 2 * (r - 4));
}

// LDS byte offset of a __shared__ object (generic AS pointer low bits = LDS offset)
static __device__ __forceinline__ unsigned lds_off(const void* p) {
    return (unsigned)(unsigned long long)p;
}
// Async copy 16B global -> LDS (CDNA5 GLOBAL_LOAD_ASYNC_TO_LDS_B128, ASYNCcnt)
static __device__ __forceinline__ void async_b128(unsigned lds, const void* g) {
    unsigned long long ga = (unsigned long long)g;
    asm volatile("global_load_async_to_lds_b128 %0, %1, off"
                 :: "v"(lds), "v"(ga) : "memory");
}
static __device__ __forceinline__ void async_b128_off16(unsigned lds, const void* g) {
    unsigned long long ga = (unsigned long long)g;
    asm volatile("global_load_async_to_lds_b128 %0, %1, off offset:16"
                 :: "v"(lds), "v"(ga) : "memory");
}
static __device__ __forceinline__ void wait_async0() {
    asm volatile("s_wait_asynccnt 0x0" ::: "memory");
}

// ---------------------------------------------------------------------------
// fp32 -> bf16 bulk convert (8 elements / thread, b128 in / b128 out)
// ---------------------------------------------------------------------------
__global__ __launch_bounds__(256) void cvt_kernel(const float* __restrict__ src,
                                                  unsigned short* __restrict__ dst)
{
    size_t i = (size_t)blockIdx.x * 256 + threadIdx.x;
    const float4* s = (const float4*)src;
    float4 a = s[2 * i], b = s[2 * i + 1];
    U4 d;
    d.x = (unsigned)f2bf(a.x) | ((unsigned)f2bf(a.y) << 16);
    d.y = (unsigned)f2bf(a.z) | ((unsigned)f2bf(a.w) << 16);
    d.z = (unsigned)f2bf(b.x) | ((unsigned)f2bf(b.y) << 16);
    d.w = (unsigned)f2bf(b.z) | ((unsigned)f2bf(b.w) << 16);
    ((U4*)dst)[i] = d;
}

// ---------------------------------------------------------------------------
// Channel-mixing GEMM:  out[m,nw] = sum_k W[m,k] * X[k,nw] + bias[m] (+resid)
// All-bf16 operands. Block tile 128(M) x 256(N), K-step 32, 8 waves (2x4),
// each wave owns a 64x64 patch = 4x4 WMMA accumulators.
// W tile staged via async global->LDS b128; X tile staged transposed.
// ---------------------------------------------------------------------------
#define GM 128
#define GN 256
#define GK 32
#define LP 40   // padded LDS row stride (bf16): row starts 16B aligned, no bank conflicts

template <bool OUTF32, bool RES>
__global__ __launch_bounds__(256) void gemm_wmma(
    const unsigned short* __restrict__ Wb,   // bf16 [M,K]
    const unsigned short* __restrict__ Xb,   // bf16 [K,N] per batch
    const float* __restrict__ bias,
    const float* __restrict__ resid,
    float* __restrict__ outF, unsigned short* __restrict__ outH,
    int Mdim, int Kdim, int Ndim)
{
    __shared__ unsigned short sW[GM * LP];   // W tile, row-major [m][k]
    __shared__ unsigned short sX[GN * LP];   // X tile, transposed [nw][k]

    const int z = blockIdx.z;                         // batch
    const size_t xoff = (size_t)z * Kdim * Ndim;
    const size_t ooff = (size_t)z * Mdim * Ndim;

    const int tid   = threadIdx.x;
    const int lane  = tid & 31;
    const int wid   = tid >> 5;
    const int waveM = wid & 1;        // 0..1  -> 64 rows each
    const int waveN = wid >> 1;       // 0..3  -> 64 cols each
    const int bm0   = blockIdx.y * GM;
    const int bn0   = blockIdx.x * GN;
    const int mlane = lane & 15;

    // per-thread staging geometry
    const int wrow = tid >> 1;            // 0..127
    const int wcol = (tid & 1) << 4;      // 0 or 16
    const unsigned ldsW = lds_off(&sW[wrow * LP + wcol]);
    const unsigned short* wgp0 = Wb + (size_t)(bm0 + wrow) * Kdim + wcol;
    const unsigned short* xgp0 = Xb + xoff + bn0 + tid;

    v8f acc[4][4];
#pragma unroll
    for (int i = 0; i < 4; ++i)
#pragma unroll
        for (int j = 0; j < 4; ++j) acc[i][j] = vzero8();

    for (int k0 = 0; k0 < Kdim; k0 += GK) {
        __syncthreads();                       // previous iter's reads done
        // ---- W tile: pure async global->LDS (2 x b128 per thread) ----
        async_b128(ldsW, wgp0 + k0);
        async_b128_off16(ldsW, wgp0 + k0);
        // ---- X tile: 32 strided u16 -> packed -> 4 x ds_store_b128 ----
        {
            const unsigned short* xp = xgp0 + (size_t)k0 * Ndim;
            unsigned dw[16];
#pragma unroll
            for (int k2 = 0; k2 < 16; ++k2) {
                unsigned lo = xp[(size_t)(2 * k2) * Ndim];
                unsigned hi = xp[(size_t)(2 * k2 + 1) * Ndim];
                dw[k2] = lo | (hi << 16);
            }
            U4* dst = (U4*)&sX[tid * LP];
            dst[0] = U4{dw[0],  dw[1],  dw[2],  dw[3]};
            dst[1] = U4{dw[4],  dw[5],  dw[6],  dw[7]};
            dst[2] = U4{dw[8],  dw[9],  dw[10], dw[11]};
            dst[3] = U4{dw[12], dw[13], dw[14], dw[15]};
            if (k0 + GK < Kdim) {
                __builtin_prefetch(xp + (size_t)GK * Ndim, 0, 0);
                __builtin_prefetch(wgp0 + k0 + GK, 0, 0);
            }
        }
        wait_async0();
        __syncthreads();

        // ---- A fragments (one per m-tile) ----
        FragU afr[4];
#pragma unroll
        for (int mt = 0; mt < 4; ++mt) {
            const unsigned short* ap = &sW[(waveM * 64 + mt * 16 + mlane) * LP];
#pragma unroll
            for (int r = 0; r < 8; ++r)
                afr[mt].u[r] = *(const unsigned*)(ap + frag_k0(r, lane));
        }
        // ---- B fragments + WMMA ----
#pragma unroll
        for (int nt = 0; nt < 4; ++nt) {
            FragU bfr;
            const unsigned short* bp = &sX[(waveN * 64 + nt * 16 + mlane) * LP];
#pragma unroll
            for (int r = 0; r < 8; ++r)
                bfr.u[r] = *(const unsigned*)(bp + frag_k0(r, lane));
#pragma unroll
            for (int mt = 0; mt < 4; ++mt)
                acc[mt][nt] = __builtin_amdgcn_wmma_f32_16x16x32_bf16(
                    false, afr[mt].v, false, bfr.v, (short)0, acc[mt][nt],
                    false, false);
        }
    }

    // ---- epilogue: bias (+residual), store ----
    const int mhalf = (lane >> 4) << 3;   // 0 or 8
#pragma unroll
    for (int mt = 0; mt < 4; ++mt) {
        int mbase = bm0 + waveM * 64 + mt * 16 + mhalf;
#pragma unroll
        for (int nt = 0; nt < 4; ++nt) {
            int gcol = bn0 + waveN * 64 + nt * 16 + mlane;
#pragma unroll
            for (int r = 0; r < 8; ++r) {
                int grow = mbase + r;
                float v = acc[mt][nt][r] + bias[grow];
                size_t idx = ooff + (size_t)grow * Ndim + gcol;
                if (RES) v += resid[idx];
                if (OUTF32) outF[idx] = v;
                else        outH[idx] = f2bf(v);
            }
        }
    }
}

// ---------------------------------------------------------------------------
// Softmax over the position axis (row length 8192), one block per row.
// ---------------------------------------------------------------------------
__global__ __launch_bounds__(256) void softmax_w_kernel(unsigned short* buf, int Wn)
{
    __shared__ float red[256];
    const int tid = threadIdx.x;
    unsigned short* p = buf + (size_t)blockIdx.x * Wn;

    float v[32];
    float mx = -3.4e38f;
#pragma unroll
    for (int j = 0; j < 32; ++j) {
        v[j] = bf2f(p[tid + j * 256]);
        mx = fmaxf(mx, v[j]);
    }
    red[tid] = mx; __syncthreads();
    for (int s = 128; s > 0; s >>= 1) {
        if (tid < s) red[tid] = fmaxf(red[tid], red[tid + s]);
        __syncthreads();
    }
    float rowmax = red[0];
    __syncthreads();

    float sum = 0.f;
#pragma unroll
    for (int j = 0; j < 32; ++j) {
        v[j] = __expf(v[j] - rowmax);
        sum += v[j];
    }
    red[tid] = sum; __syncthreads();
    for (int s = 128; s > 0; s >>= 1) {
        if (tid < s) red[tid] += red[tid + s];
        __syncthreads();
    }
    float inv = 1.f / red[0];
#pragma unroll
    for (int j = 0; j < 32; ++j)
        p[tid + j * 256] = f2bf(v[j] * inv);
}

// ---------------------------------------------------------------------------
// Softmax over the 64 head-channels at each (n,h,w).
// ---------------------------------------------------------------------------
__global__ __launch_bounds__(256) void softmax_c_kernel(unsigned short* buf)
{
    const int tid = threadIdx.x;
    const int w   = blockIdx.x * 256 + tid;
    const int nb  = blockIdx.y >> 3;
    const int h   = blockIdx.y & 7;
    unsigned short* base = buf + ((size_t)nb * 512 + h * 64) * 8192 + w;

    float v[64];
    float mx = -3.4e38f;
#pragma unroll
    for (int c = 0; c < 64; ++c) {
        v[c] = bf2f(base[(size_t)c * 8192]);
        mx = fmaxf(mx, v[c]);
    }
    float sum = 0.f;
#pragma unroll
    for (int c = 0; c < 64; ++c) {
        v[c] = __expf(v[c] - mx);
        sum += v[c];
    }
    float inv = 1.f / sum;
#pragma unroll
    for (int c = 0; c < 64; ++c)
        base[(size_t)c * 8192] = f2bf(v[c] * inv);
}

// ---------------------------------------------------------------------------
// ctx[n,h] = K(64 x 8192) * V^T -> 64x64 fp32. One block per (n,h).
// Each wave owns 2 of the 16 output tiles and streams the full w range.
// ---------------------------------------------------------------------------
__global__ __launch_bounds__(256) void ctx_kernel(
    const unsigned short* __restrict__ Kb,
    const unsigned short* __restrict__ Vb,
    float* __restrict__ ctx)
{
    const int z  = blockIdx.x;          // n*H
    const int nb = z >> 3, h = z & 7;
    const unsigned short* Kp = Kb + ((size_t)nb * 512 + h * 64) * 8192;
    const unsigned short* Vp = Vb + ((size_t)nb * 512 + h * 64) * 8192;

    const int lane  = threadIdx.x & 31;
    const int wid   = threadIdx.x >> 5;
    const int mlane = lane & 15;
    const int mt  = wid >> 1;
    const int nt0 = (2 * wid) & 3;
    const int nt1 = nt0 + 1;

    v8f a0 = vzero8(), a1 = vzero8();

    const unsigned short* ap0 = Kp + (size_t)(mt * 16 + mlane) * 8192;
    const unsigned short* bp0 = Vp + (size_t)(nt0 * 16 + mlane) * 8192;
    const unsigned short* bp1 = Vp + (size_t)(nt1 * 16 + mlane) * 8192;

    for (int w0 = 0; w0 < 8192; w0 += 32) {
        FragU a, b0, b1;
#pragma unroll
        for (int r = 0; r < 8; ++r) {
            int kk = frag_k0(r, lane);
            a.u[r]  = *(const unsigned*)(ap0 + w0 + kk);
            b0.u[r] = *(const unsigned*)(bp0 + w0 + kk);
            b1.u[r] = *(const unsigned*)(bp1 + w0 + kk);
        }
        a0 = __builtin_amdgcn_wmma_f32_16x16x32_bf16(false, a.v, false, b0.v,
                                                     (short)0, a0, false, false);
        a1 = __builtin_amdgcn_wmma_f32_16x16x32_bf16(false, a.v, false, b1.v,
                                                     (short)0, a1, false, false);
    }

    float* cp = ctx + (size_t)z * 64 * 64;
    const int mbase = mt * 16 + ((lane >> 4) << 3);
#pragma unroll
    for (int r = 0; r < 8; ++r) {
        cp[(size_t)(mbase + r) * 64 + nt0 * 16 + mlane] = a0[r];
        cp[(size_t)(mbase + r) * 64 + nt1 * 16 + mlane] = a1[r];
    }
}

// ---------------------------------------------------------------------------
// att[n,h] = ctx^T (64x64) * Q (64 x 8192).  grid = (8192/256, n*H).
// ---------------------------------------------------------------------------
__global__ __launch_bounds__(256) void att_kernel(
    const float* __restrict__ ctx,
    const unsigned short* __restrict__ Qb,
    unsigned short* __restrict__ attb)
{
    __shared__ float sctx[64 * 64];

    const int z  = blockIdx.y;
    const int nb = z >> 3, h = z & 7;
    const int tid = threadIdx.x;
    const int lane = tid & 31;
    const int wid  = tid >> 5;
    const int mlane = lane & 15;

    const float* cp = ctx + (size_t)z * 64 * 64;
    for (int i = tid; i < 64 * 64; i += 256) sctx[i] = cp[i];
    __syncthreads();

    const unsigned short* Qp = Qb + ((size_t)nb * 512 + h * 64) * 8192;
    unsigned short*       Ap = attb + ((size_t)nb * 512 + h * 64) * 8192;
    const int w0 = blockIdx.x * 256 + wid * 32;

    v8f acc[4][2];
#pragma unroll
    for (int mt = 0; mt < 4; ++mt) { acc[mt][0] = vzero8(); acc[mt][1] = vzero8(); }

#pragma unroll
    for (int kt = 0; kt < 2; ++kt) {
        const int kbase = kt * 32;
        FragU afr[4];
#pragma unroll
        for (int mt = 0; mt < 4; ++mt) {
            int m = mt * 16 + mlane;       // column of ctx (= row of ctx^T)
#pragma unroll
            for (int r = 0; r < 8; ++r) {
                int k = kbase + frag_k0(r, lane);
                unsigned lo = f2bf(sctx[(size_t)k * 64 + m]);
                unsigned hi = f2bf(sctx[(size_t)(k + 1) * 64 + m]);
                afr[mt].u[r] = lo | (hi << 16);
            }
        }
#pragma unroll
        for (int ntw = 0; ntw < 2; ++ntw) {
            FragU bfr;
            int wcol = w0 + ntw * 16 + mlane;
#pragma unroll
            for (int r = 0; r < 8; ++r) {
                int k = kbase + frag_k0(r, lane);
                unsigned lo = Qp[(size_t)k * 8192 + wcol];
                unsigned hi = Qp[(size_t)(k + 1) * 8192 + wcol];
                bfr.u[r] = lo | (hi << 16);
            }
#pragma unroll
            for (int mt = 0; mt < 4; ++mt)
                acc[mt][ntw] = __builtin_amdgcn_wmma_f32_16x16x32_bf16(
                    false, afr[mt].v, false, bfr.v, (short)0, acc[mt][ntw],
                    false, false);
        }
    }

    const int mhalf = (lane >> 4) << 3;
#pragma unroll
    for (int mt = 0; mt < 4; ++mt) {
        int vbase = mt * 16 + mhalf;
#pragma unroll
        for (int ntw = 0; ntw < 2; ++ntw) {
            int wcol = w0 + ntw * 16 + mlane;
#pragma unroll
            for (int r = 0; r < 8; ++r)
                Ap[(size_t)(vbase + r) * 8192 + wcol] = f2bf(acc[mt][ntw][r]);
        }
    }
}

// ---------------------------------------------------------------------------
// Host-side launch
// ---------------------------------------------------------------------------
extern "C" void kernel_launch(void* const* d_in, const int* in_sizes, int n_in,
                              void* d_out, int out_size, void* d_ws, size_t ws_size,
                              hipStream_t stream)
{
    (void)in_sizes; (void)n_in; (void)out_size; (void)ws_size;

    const float* x  = (const float*)d_in[0];
    const float* Wk = (const float*)d_in[1];
    const float* bk = (const float*)d_in[2];
    const float* Wq = (const float*)d_in[3];
    const float* bq = (const float*)d_in[4];
    const float* Wv = (const float*)d_in[5];
    const float* bv = (const float*)d_in[6];
    const float* Wr = (const float*)d_in[7];
    const float* br = (const float*)d_in[8];
    float* out = (float*)d_out;

    const size_t elems = 8ull * 512ull * 8192ull;   // 33,554,432
    const size_t welems = 512ull * 512ull;          // 262,144
    unsigned short* bufK = (unsigned short*)d_ws;   // 64 MB each
    unsigned short* bufQ = bufK + elems;
    unsigned short* bufV = bufQ + elems;
    unsigned short* bufX = bufV + elems;            // bf16 copy of x
    unsigned short* wKb  = bufX + elems;            // bf16 weights, 512 KB each
    unsigned short* wQb  = wKb + welems;
    unsigned short* wVb  = wQb + welems;
    unsigned short* wRb  = wVb + welems;
    float*          ctxb = (float*)(wRb + welems);  // 1 MB

    // 0) one-time bf16 conversion of x and weights
    cvt_kernel<<<(unsigned)(elems / (256 * 8)), 256, 0, stream>>>(x, bufX);
    cvt_kernel<<<(unsigned)(welems / (256 * 8)), 256, 0, stream>>>(Wk, wKb);
    cvt_kernel<<<(unsigned)(welems / (256 * 8)), 256, 0, stream>>>(Wq, wQb);
    cvt_kernel<<<(unsigned)(welems / (256 * 8)), 256, 0, stream>>>(Wv, wVb);
    cvt_kernel<<<(unsigned)(welems / (256 * 8)), 256, 0, stream>>>(Wr, wRb);

    dim3 gg(8192 / GN, 512 / GM, 8);   // (32, 4, 8)

    // 1) projections (bf16 GEMM, bias fused, bf16 out)
    gemm_wmma<false, false><<<gg, 256, 0, stream>>>(
        wKb, bufX, bk, nullptr, nullptr, bufK, 512, 512, 8192);
    gemm_wmma<false, false><<<gg, 256, 0, stream>>>(
        wQb, bufX, bq, nullptr, nullptr, bufQ, 512, 512, 8192);
    gemm_wmma<false, false><<<gg, 256, 0, stream>>>(
        wVb, bufX, bv, nullptr, nullptr, bufV, 512, 512, 8192);

    // 2) softmaxes (in place)
    softmax_w_kernel<<<8 * 512, 256, 0, stream>>>(bufK, 8192);
    softmax_c_kernel<<<dim3(8192 / 256, 8 * 8), 256, 0, stream>>>(bufQ);

    // 3) ctx = K V^T  (per head, 64x64 fp32)
    ctx_kernel<<<8 * 8, 256, 0, stream>>>(bufK, bufV, ctxb);

    // 4) att = ctx^T Q  (bf16, reuses the K buffer)
    att_kernel<<<dim3(8192 / 256, 8 * 8), 256, 0, stream>>>(ctxb, bufQ, bufK);

    // 5) out = Wr att + br + x  (fp32 out, residual fused)
    gemm_wmma<true, true><<<gg, 256, 0, stream>>>(
        wRb, bufK, br, x, out, nullptr, 512, 512, 8192);
}
